// GroupedQueryAttentionRoPE_29996051595748
// MI455X (gfx1250) — matrix operand order repaired and checked by reading, not compiled, not run on hardware
//
#include <hip/hip_runtime.h>
#include <hip/hip_bf16.h>
#include <stdint.h>

typedef __attribute__((ext_vector_type(16))) __bf16 v16bf;
typedef __attribute__((ext_vector_type(8)))  __bf16 v8bf;
typedef __attribute__((ext_vector_type(4)))  __bf16 v4bf;
typedef __attribute__((ext_vector_type(8)))  float  v8f;

#define NH     32
#define NKV    8
#define DK     64
#define TSEQ   2048
#define DMODEL 2048
#define NEG_INF (-3.402823466e38f)

static __device__ __forceinline__ v16bf cat8(v8bf lo, v8bf hi) {
  return __builtin_shufflevector(lo, hi, 0,1,2,3,4,5,6,7,8,9,10,11,12,13,14,15);
}

static __device__ __forceinline__ v8f wmma_bf16(v16bf a, v16bf b, v8f c) {
  return __builtin_amdgcn_wmma_f32_16x16x32_bf16(false, a, false, b, (short)0, c, false, false);
}

// Async global->LDS 16B copy (CDNA5 ASYNCcnt path; bypasses VGPRs).
static __device__ __forceinline__ void async_copy_b128(const void* gp, void* lp) {
  unsigned lds = (unsigned)(uintptr_t)lp;        // low 32 bits of flat LDS addr = LDS offset
  unsigned long long ga = (unsigned long long)(uintptr_t)gp;
  asm volatile("global_load_async_to_lds_b128 %0, %1, off" :: "v"(lds), "v"(ga) : "memory");
}
static __device__ __forceinline__ void async_wait0() {
  asm volatile("s_wait_asynccnt 0x0" ::: "memory");
}

// ---------------------------------------------------------------------------
// Tiled bf16-WMMA GEMM: C[M,N] = A[M,K] @ W[K,N]
// MODE 0: Q projection  (fp32 A, RoPE epilogue, scatter -> [b][32][t][64] bf16)
// MODE 1: K projection  (fp32 A, RoPE epilogue, scatter -> [b][8][t][64] bf16)
// MODE 2: V projection  (fp32 A, scatter TRANSPOSED -> [b][8][64][t] bf16)
// MODE 3: out projection (bf16 A via async-to-LDS, fp32 row-major output)
// Block tile 128x64, 256 threads (8 waves), wave tile 32x32, BK=32,
// double-buffered LDS staging.
// ---------------------------------------------------------------------------
template<int MODE>
__global__ __launch_bounds__(256)
void gemm_wmma(const void* __restrict__ Aptr, const float* __restrict__ W,
               void* __restrict__ Out, int M, int N, int K) {
  __shared__ alignas(16) __bf16 As[2][128][32];   // [buf][m][k]
  __shared__ alignas(16) __bf16 Bt[2][64][32];    // [buf][n][k] (transposed W tile)

  const int tid  = threadIdx.x;
  const int lane = tid & 31;
  const int wid  = tid >> 5;
  const int wr   = wid & 3;          // wave row (4)
  const int wc   = wid >> 2;         // wave col (2)
  const int hi   = lane >> 4;        // lane half (0/1)
  const int ln   = lane & 15;

  const int m0 = blockIdx.y * 128;
  const int n0 = blockIdx.x * 64;

  v8f acc[2][2] = {};

  auto stage = [&](int k0, int buf) {
    if (MODE == 3) {
      const __bf16* A = (const __bf16*)Aptr;
      for (int idx = tid; idx < 128 * 4; idx += 256) {
        int r = idx >> 2, c = (idx & 3) * 8;
        async_copy_b128(&A[(size_t)(m0 + r) * K + k0 + c], &As[buf][r][c]);
      }
    } else {
      const float* A = (const float*)Aptr;
      for (int idx = tid; idx < 128 * 8; idx += 256) {
        int r = idx >> 3, c = (idx & 7) * 4;
        float4 f = *(const float4*)&A[(size_t)(m0 + r) * K + k0 + c];
        v4bf h = { (__bf16)f.x, (__bf16)f.y, (__bf16)f.z, (__bf16)f.w };
        *(v4bf*)&As[buf][r][c] = h;
      }
    }
    // W tile transposed: Bt[n][k] = bf16(W[k0+k][n0+n]); reads coalesced along n
    for (int idx = tid; idx < 32 * 64; idx += 256) {
      int kk = idx >> 6, nn = idx & 63;
      Bt[buf][nn][kk] = (__bf16)W[(size_t)(k0 + kk) * N + n0 + nn];
    }
  };

  const int nk = K / 32;
  stage(0, 0);
  if (MODE == 3) async_wait0();
  __syncthreads();

  for (int kt = 0; kt < nk; ++kt) {
    const int buf = kt & 1;
    if (kt + 1 < nk) stage((kt + 1) * 32, buf ^ 1);   // prefetch next tile

    // A 16x32 bf16 layout: lanes 0-15 row=ln hold K{0-15}; lanes 16-31 K{16-31}
    v16bf afr[2], bfr[2];
    for (int tr = 0; tr < 2; ++tr) {
      const v8bf* p = (const v8bf*)&As[buf][wr * 32 + tr * 16 + ln][0];
      afr[tr] = cat8(p[2 * hi], p[2 * hi + 1]);
    }
    // B 32x16 bf16 layout: lane col=ln; 16 consecutive K starting at hi*16
    for (int tc = 0; tc < 2; ++tc) {
      const v8bf* p = (const v8bf*)&Bt[buf][wc * 32 + tc * 16 + ln][0];
      bfr[tc] = cat8(p[2 * hi], p[2 * hi + 1]);
    }
    for (int tr = 0; tr < 2; ++tr)
      for (int tc = 0; tc < 2; ++tc)
        acc[tr][tc] = wmma_bf16(afr[tr], bfr[tc], acc[tr][tc]);

    if (MODE == 3) async_wait0();
    __syncthreads();
  }

  // ---- epilogue: C fragment row = e + hi*8, col = ln (within 16x16 tile) ----
  for (int tr = 0; tr < 2; ++tr) {
    for (int tc = 0; tc < 2; ++tc) {
      const int n = n0 + wc * 32 + tc * 16 + ln;
      const int d = n & (DK - 1);
      float freq = 0.f;
      if (MODE == 0 || MODE == 1) {    // hoisted: frequency depends only on column
        const int i2 = d & ~1;
        freq = __expf(-(float)i2 * (9.210340371976184f / 64.0f));
      }
      for (int e = 0; e < 8; ++e) {
        const int m = m0 + wr * 32 + tr * 16 + e + hi * 8;
        float x = acc[tr][tc][e];
        if (MODE == 3) {
          ((float*)Out)[(size_t)m * N + n] = x;
        } else {
          const int b = m / TSEQ, t = m % TSEQ;
          float val = x;
          if (MODE == 0 || MODE == 1) {
            // RoPE: even/odd head-dim columns sit in adjacent lanes
            float partner = __shfl_xor(x, 1);
            float ang = (float)t * freq;
            float cs = __cosf(ang), sn = __sinf(ang);
            val = (d & 1) ? (partner * sn + x * cs)        // odd:  x1*sin + x2*cos
                          : (x * cs - partner * sn);       // even: x1*cos - x2*sin
          }
          const int h = n >> 6;
          if (MODE == 2) {             // V stored transposed per kv-head: [d][t]
            ((__bf16*)Out)[(((size_t)b * NKV + h) * DK + d) * TSEQ + t] = (__bf16)val;
          } else {
            const int nheads = (MODE == 0) ? NH : NKV;
            ((__bf16*)Out)[(((size_t)b * nheads + h) * TSEQ + t) * DK + d] = (__bf16)val;
          }
        }
      }
    }
  }
}

// ---------------------------------------------------------------------------
// Flash-style GQA attention. Block = 128 threads (4 waves), one (b,h), 64 q rows.
// Each wave owns 16 q rows; K/V streamed through double-buffered LDS in 64-key
// blocks via async global->LDS copies (V pre-transposed in global memory).
// ---------------------------------------------------------------------------
__global__ __launch_bounds__(128)
void attn_wmma(const void* __restrict__ Qp, const void* __restrict__ Kp,
               const void* __restrict__ Vp, void* __restrict__ Op) {
  __shared__ alignas(16) __bf16 Ks[2][64][64];     // [buf][key][d]
  __shared__ alignas(16) __bf16 Vt[2][64][64];     // [buf][d][key]
  __shared__ alignas(16) __bf16 Ps[4][16][64];     // per-wave P tile [qrow][key]

  const __bf16* Q  = (const __bf16*)Qp;
  const __bf16* Kb = (const __bf16*)Kp;
  const __bf16* Vb = (const __bf16*)Vp;
  __bf16*       O  = (__bf16*)Op;

  const int tid = threadIdx.x, lane = tid & 31, wid = tid >> 5;
  const int hi = lane >> 4, ln = lane & 15;
  const int b = blockIdx.z, h = blockIdx.y;
  const int q0 = blockIdx.x * 64 + wid * 16;
  const int kvh = h >> 2;                          // G = 32/8 = 4
  const float scale = 0.125f;                      // 1/sqrt(64)

  // Q A-fragments for this wave's 16 rows (d 0..31 and 32..63)
  const v8bf* qp = (const v8bf*)&Q[(((size_t)b * NH + h) * TSEQ + q0 + ln) * DK];
  v16bf qa0 = cat8(qp[2 * hi],     qp[2 * hi + 1]);
  v16bf qa1 = cat8(qp[4 + 2 * hi], qp[4 + 2 * hi + 1]);

  const __bf16* Kbase = &Kb[((size_t)b * NKV + kvh) * TSEQ * DK];   // [key][d]
  const __bf16* Vbase = &Vb[((size_t)b * NKV + kvh) * DK * TSEQ];   // [d][t]

  v8f acc[4] = {};
  float mrow[8], lrow[8], f[8];
  for (int e = 0; e < 8; ++e) { mrow[e] = NEG_INF; lrow[e] = 0.f; }

  auto stage = [&](int kb, int buf) {
    for (int idx = tid; idx < 64 * 8; idx += 128) {
      int r = idx >> 3, c = (idx & 7) * 8;
      async_copy_b128(&Kbase[(size_t)(kb * 64 + r) * DK + c], &Ks[buf][r][c]);
      async_copy_b128(&Vbase[(size_t)r * TSEQ + kb * 64 + c], &Vt[buf][r][c]);
    }
  };

  const int NB = TSEQ / 64;
  stage(0, 0);
  async_wait0();
  __syncthreads();

  for (int kb = 0; kb < NB; ++kb) {
    const int buf = kb & 1;
    if (kb + 1 < NB) stage(kb + 1, buf ^ 1);       // async prefetch next K/V block

    // ---- scores: 4 key tiles of 16, K-dim = 64 (2 WMMAs each) ----
    v8f s[4];
    for (int kt = 0; kt < 4; ++kt) {
      const v8bf* kp = (const v8bf*)&Ks[buf][kt * 16 + ln][0];  // B col = key, K = d
      v16bf b0 = cat8(kp[2 * hi],     kp[2 * hi + 1]);
      v16bf b1 = cat8(kp[4 + 2 * hi], kp[4 + 2 * hi + 1]);
      v8f z = {};
      z = wmma_bf16(qa0, b0, z);
      z = wmma_bf16(qa1, b1, z);
      s[kt] = z;
    }

    // ---- online softmax (row reductions across 16-lane halves) ----
    for (int e = 0; e < 8; ++e) {
      float bm = fmaxf(fmaxf(s[0][e], s[1][e]), fmaxf(s[2][e], s[3][e])) * scale;
      for (int msk = 1; msk < 16; msk <<= 1) bm = fmaxf(bm, __shfl_xor(bm, msk));
      float nm = fmaxf(mrow[e], bm);
      f[e] = __expf(mrow[e] - nm);
      mrow[e] = nm;
    }
    float rs[8] = {0.f, 0.f, 0.f, 0.f, 0.f, 0.f, 0.f, 0.f};
    for (int kt = 0; kt < 4; ++kt)
      for (int e = 0; e < 8; ++e) {
        float p = __expf(s[kt][e] * scale - mrow[e]);
        rs[e] += p;
        Ps[wid][e + hi * 8][kt * 16 + ln] = (__bf16)p;
      }
    for (int e = 0; e < 8; ++e) {
      float sum = rs[e];
      for (int msk = 1; msk < 16; msk <<= 1) sum += __shfl_xor(sum, msk);
      lrow[e] = lrow[e] * f[e] + sum;
    }
    for (int dt = 0; dt < 4; ++dt)
      for (int e = 0; e < 8; ++e) acc[dt][e] *= f[e];

    __syncthreads();   // Ps visible across lanes of the wave

    // ---- P @ V: A = P (16 x 64 keys), B = Vt slices ----
    const v8bf* pp = (const v8bf*)&Ps[wid][ln][0];
    v16bf pa0 = cat8(pp[2 * hi],     pp[2 * hi + 1]);
    v16bf pa1 = cat8(pp[4 + 2 * hi], pp[4 + 2 * hi + 1]);
    for (int dt = 0; dt < 4; ++dt) {
      const v8bf* vp = (const v8bf*)&Vt[buf][dt * 16 + ln][0];
      v16bf vb0 = cat8(vp[2 * hi],     vp[2 * hi + 1]);
      v16bf vb1 = cat8(vp[4 + 2 * hi], vp[4 + 2 * hi + 1]);
      acc[dt] = wmma_bf16(pa0, vb0, acc[dt]);
      acc[dt] = wmma_bf16(pa1, vb1, acc[dt]);
    }

    async_wait0();     // drain prefetch before it becomes the active buffer
    __syncthreads();   // and before next stage overwrites buf
  }

  // ---- normalize and write [b][t][h*64+d] bf16 for the output projection ----
  for (int dt = 0; dt < 4; ++dt)
    for (int e = 0; e < 8; ++e) {
      int t = q0 + e + hi * 8;
      float val = acc[dt][e] / lrow[e];
      O[((size_t)b * TSEQ + t) * DMODEL + h * DK + dt * 16 + ln] = (__bf16)val;
    }
}

// ---------------------------------------------------------------------------
extern "C" void kernel_launch(void* const* d_in, const int* in_sizes, int n_in,
                              void* d_out, int out_size, void* d_ws, size_t ws_size,
                              hipStream_t stream) {
  const float* q  = (const float*)d_in[0];
  const float* k  = (const float*)d_in[1];
  const float* v  = (const float*)d_in[2];
  const float* Wq = (const float*)d_in[3];
  const float* Wk = (const float*)d_in[4];
  const float* Wv = (const float*)d_in[5];
  const float* Wo = (const float*)d_in[6];

  char* ws = (char*)d_ws;
  void* Qws = (void*)(ws);                                   // 16 MB: [2][32][2048][64] bf16
  void* Kws = (void*)(ws + (size_t)16 * 1024 * 1024);        //  4 MB: [2][8][2048][64]  bf16
  void* Vws = (void*)(ws + (size_t)20 * 1024 * 1024);        //  4 MB: [2][8][64][2048]  bf16 (transposed)
  void* Aws = (void*)(ws + (size_t)24 * 1024 * 1024);        // 16 MB: [4096][2048] bf16

  const int M = 2 * TSEQ;  // 4096 rows
  dim3 blk(256);

  gemm_wmma<0><<<dim3(DMODEL / 64,     M / 128), blk, 0, stream>>>(q, Wq, Qws, M, DMODEL,   DMODEL);
  gemm_wmma<1><<<dim3((NKV * DK) / 64, M / 128), blk, 0, stream>>>(k, Wk, Kws, M, NKV * DK, DMODEL);
  gemm_wmma<2><<<dim3((NKV * DK) / 64, M / 128), blk, 0, stream>>>(v, Wv, Vws, M, NKV * DK, DMODEL);

  attn_wmma<<<dim3(TSEQ / 64, NH, 2), dim3(128), 0, stream>>>(Qws, Kws, Vws, Aws);

  gemm_wmma<3><<<dim3(DMODEL / 64, M / 128), blk, 0, stream>>>(Aws, Wo, d_out, M, DMODEL, DMODEL);
}
